// PointNetSetAbstractionMsg_68496138437064
// MI455X (gfx1250) — matrix operand order repaired and checked by reading, not compiled, not run on hardware
//
#include <hip/hip_runtime.h>
#include <hip/hip_bf16.h>

typedef __attribute__((ext_vector_type(16))) _Float16 v16h;
typedef __attribute__((ext_vector_type(8)))  _Float16 v8h;
typedef __attribute__((ext_vector_type(8)))  float    v8f;

#define B_    8
#define N_    4096
#define S_    1024
#define CIN_  64
#define CTOT_ 320

#define CAT16(a, b) __builtin_shufflevector(a, b, 0,1,2,3,4,5,6,7,8,9,10,11,12,13,14,15)

__device__ __forceinline__ v8f wmma32(v16h a, v16h b, v8f c) {
  return __builtin_amdgcn_wmma_f32_16x16x32_f16(false, a, false, b, (short)0, c, false, false);
}

// ---------------------------------------------------------------------------
// FPS: one block per (batch, run). 1024 threads, 4 points each (strided).
// Emits idx[t] = far BEFORE update (matches jax scan emitting incoming `far`).
// ---------------------------------------------------------------------------
__global__ __launch_bounds__(1024) void fps_kernel(
    const float* __restrict__ xyz,      // (B,3,N)
    const int* __restrict__ init1, const int* __restrict__ init2,
    int* __restrict__ out1, int* __restrict__ out2)
{
  const int b = blockIdx.x;
  const int run = blockIdx.y;
  const int* initp = run == 0 ? init1 : init2;
  int* outp = run == 0 ? out1 : out2;
  const float* px = xyz + (long)b * 3 * N_;
  const float* py = px + N_;
  const float* pz = px + 2 * N_;
  const int tid = threadIdx.x;

  float lx[4], ly[4], lz[4], dmin[4];
#pragma unroll
  for (int j = 0; j < 4; ++j) {
    int n = tid + j * 1024;
    lx[j] = px[n]; ly[j] = py[n]; lz[j] = pz[n];
    dmin[j] = 1e10f;
  }

  __shared__ float sVal[32];
  __shared__ int   sIdx[32];
  __shared__ int   sFar;

  int far = initp[b];
  for (int t = 0; t < S_; ++t) {
    if (tid == 0) outp[(long)b * S_ + t] = far;
    float cx = px[far], cy = py[far], cz = pz[far];
    float bestV = -1.0f; int bestI = 0x7fffffff;
#pragma unroll
    for (int j = 0; j < 4; ++j) {
      int n = tid + j * 1024;
      float dx = lx[j] - cx, dy = ly[j] - cy, dz = lz[j] - cz;
      float d = dx * dx + dy * dy + dz * dz;
      dmin[j] = fminf(dmin[j], d);
      float v = dmin[j];
      if (v > bestV || (v == bestV && n < bestI)) { bestV = v; bestI = n; }
    }
    // wave32 reduce (first-max tie-break = lowest index)
    for (int off = 16; off > 0; off >>= 1) {
      float ov = __shfl_down(bestV, off, 32);
      int   oi = __shfl_down(bestI, off, 32);
      if (ov > bestV || (ov == bestV && oi < bestI)) { bestV = ov; bestI = oi; }
    }
    if ((tid & 31) == 0) { sVal[tid >> 5] = bestV; sIdx[tid >> 5] = bestI; }
    __syncthreads();
    if (tid < 32) {
      bestV = sVal[tid]; bestI = sIdx[tid];
      for (int off = 16; off > 0; off >>= 1) {
        float ov = __shfl_down(bestV, off, 32);
        int   oi = __shfl_down(bestI, off, 32);
        if (ov > bestV || (ov == bestV && oi < bestI)) { bestV = ov; bestI = oi; }
      }
      if (tid == 0) sFar = bestI;
    }
    __syncthreads();
    far = sFar;
  }
}

// ---------------------------------------------------------------------------
// Gather new_xyz / new_color; write ws copies (S-major) and outputs (B,3,S).
// ---------------------------------------------------------------------------
__global__ void gather_new(
    const float* __restrict__ xyz, const float* __restrict__ color,
    const int* __restrict__ fidx1, const int* __restrict__ fidx2,
    float* __restrict__ nxyz_ws, float* __restrict__ ncol_ws,
    float* __restrict__ out_xyz, float* __restrict__ out_col)
{
  int i = blockIdx.x * blockDim.x + threadIdx.x;
  if (i >= B_ * S_) return;
  int b = i / S_, s = i % S_;
  int i1 = fidx1[i], i2 = fidx2[i];
#pragma unroll
  for (int c = 0; c < 3; ++c) {
    float vx = xyz[((long)b * 3 + c) * N_ + i1];
    float vc = color[((long)b * 3 + c) * N_ + i2];
    nxyz_ws[(long)i * 3 + c] = vx;
    ncol_ws[(long)i * 3 + c] = vc;
    out_xyz[((long)b * 3 + c) * S_ + s] = vx;
    out_col[((long)b * 3 + c) * S_ + s] = vc;
  }
}

// ---------------------------------------------------------------------------
// Ball query: one wave per (b,s). Stream-compact first K in-range indices
// (ascending), pad with slot 0 (matches reference sort + first-fill).
// ---------------------------------------------------------------------------
__global__ __launch_bounds__(256) void ball_query(
    const float* __restrict__ xyz, const float* __restrict__ nxyz_ws,
    int* __restrict__ gidx, float* __restrict__ gsq,
    float r2, int K)
{
  __shared__ int   sI[8][64];
  __shared__ float sD[8][64];
  const int wave = threadIdx.x >> 5, lane = threadIdx.x & 31;
  const int q = blockIdx.x * 8 + wave;         // b*S + s
  const int b = q / S_;
  const float* px = xyz + (long)b * 3 * N_;
  const float* py = px + N_;
  const float* pz = px + 2 * N_;
  const float qx = nxyz_ws[(long)q * 3 + 0];
  const float qy = nxyz_ws[(long)q * 3 + 1];
  const float qz = nxyz_ws[(long)q * 3 + 2];

  int cnt = 0;
  for (int base = 0; base < N_ && cnt < K; base += 32) {
    int n = base + lane;
    float dx = px[n] - qx, dy = py[n] - qy, dz = pz[n] - qz;
    float d = dx * dx + dy * dy + dz * dz;
    bool in = (d <= r2);
    unsigned mask = (unsigned)__ballot(in);
    int pos = cnt + __popc(mask & ((1u << lane) - 1u));
    if (in && pos < K) { sI[wave][pos] = n; sD[wave][pos] = d; }
    cnt += __popc(mask);
  }
  if (cnt > K) cnt = K;
  __syncthreads();
  int fI = sI[wave][0]; float fD = sD[wave][0];
  for (int k = lane; k < K; k += 32) {
    int   id = (k < cnt) ? sI[wave][k] : fI;
    float dd = (k < cnt) ? sD[wave][k] : fD;
    gidx[(long)q * K + k] = id;
    gsq[(long)q * K + k] = dd;
  }
}

// ---------------------------------------------------------------------------
// Build first-layer activations (M, 96) f16:
// [points(64) | grouped_xyz(3) | yuan(3) | sq(1) | zero-pad(25)]
// ---------------------------------------------------------------------------
__global__ void build_gp(
    const float* __restrict__ points, const float* __restrict__ xyz,
    const float* __restrict__ nxyz_ws,
    const int* __restrict__ gidx, const float* __restrict__ gsq,
    _Float16* __restrict__ act, int K, long M)
{
  long i = (long)blockIdx.x * blockDim.x + threadIdx.x;
  if (i >= M * 96) return;
  int c = (int)(i % 96);
  long row = i / 96;
  long q = row / K;                 // b*S + s
  int b = (int)(q / S_);
  int idx = gidx[row];
  float v;
  if (c < 64)       v = points[((long)b * CIN_ + c) * N_ + idx];
  else if (c < 67)  v = xyz[((long)b * 3 + (c - 64)) * N_ + idx] - nxyz_ws[q * 3 + (c - 64)];
  else if (c < 70)  v = nxyz_ws[q * 3 + (c - 67)];
  else if (c == 70) v = gsq[row];
  else              v = 0.0f;
  act[i] = (_Float16)v;
}

// ---------------------------------------------------------------------------
// Convert fp32 weights (Cout, CinReal) -> f16 (Cout, CinPad), zero padded.
// ---------------------------------------------------------------------------
__global__ void conv_weights(const float* __restrict__ W, _Float16* __restrict__ Wh,
                             int Cout, int CinReal, int CinPad)
{
  int i = blockIdx.x * blockDim.x + threadIdx.x;
  if (i >= Cout * CinPad) return;
  int o = i / CinPad, c = i % CinPad;
  Wh[i] = (_Float16)((c < CinReal) ? W[o * CinReal + c] : 0.0f);
}

__global__ void zero_stats(float* __restrict__ s0, float* __restrict__ s1, int n)
{
  int i = blockIdx.x * blockDim.x + threadIdx.x;
  if (i < n) { s0[i] = 0.0f; s1[i] = 0.0f; }
}

// ---------------------------------------------------------------------------
// WMMA GEMM, register-tiled: each wave computes a 32-row x (NT*16)-col macro
// tile. Per K-step: 2 A-fragment loads reused across NT N-tiles; each B
// fragment reused across both M-tiles -> ~1.25 b128 loads per WMMA.
// Fused bias add + per-channel sum/sum^2 (LDS reduce, 1 atomic/ch/block).
// ---------------------------------------------------------------------------
template <int NT>
__global__ __launch_bounds__(256) void gemm_bias_stats(
    const _Float16* __restrict__ A, const _Float16* __restrict__ W,
    const float* __restrict__ bias, float* __restrict__ D,
    float* __restrict__ sum, float* __restrict__ sumsq,
    int M, int Kpad)
{
  constexpr int Nout = NT * 16;
  const int lane   = threadIdx.x & 31;
  const int wave   = threadIdx.x >> 5;
  const int half16 = lane >> 4;
  const int l16    = lane & 15;
  const int mt0    = (blockIdx.x * 8 + wave) * 2;   // two 16-row tiles per wave

  const long arow0 = (long)(mt0 * 16 + l16) * Kpad;
  const long arow1 = arow0 + (long)16 * Kpad;
  const long bbase = (long)l16 * Kpad;

  const v8f vzero = {};
  v8f acc[2][NT];
#pragma unroll
  for (int m = 0; m < 2; ++m)
#pragma unroll
    for (int nt = 0; nt < NT; ++nt) acc[m][nt] = vzero;

  for (int k = 0; k < Kpad; k += 32) {
    // A frag: lanes 0-15 -> K {0..7,16..23}; lanes 16-31 -> K {8..15,24..31}
    v8h a00 = *(const v8h*)(A + arow0 + k + half16 * 8);
    v8h a01 = *(const v8h*)(A + arow0 + k + 16 + half16 * 8);
    v8h a10 = *(const v8h*)(A + arow1 + k + half16 * 8);
    v8h a11 = *(const v8h*)(A + arow1 + k + 16 + half16 * 8);
    v16h av0 = CAT16(a00, a01);
    v16h av1 = CAT16(a10, a11);
#pragma unroll
    for (int nt = 0; nt < NT; ++nt) {
      // B frag: lanes 0-15 -> K {0..7,16..23}; lanes 16-31 -> K {8..15,24..31} (col = l16)
      const _Float16* wp = W + bbase + (long)nt * 16 * Kpad + k + half16 * 8;
      v8h b0 = *(const v8h*)(wp);
      v8h b1 = *(const v8h*)(wp + 16);
      v16h bv = CAT16(b0, b1);
      acc[0][nt] = wmma32(av0, bv, acc[0][nt]);
      acc[1][nt] = wmma32(av1, bv, acc[1][nt]);
    }
  }

  __shared__ float sSum[Nout], sSq[Nout];
  for (int i = threadIdx.x; i < Nout; i += 256) { sSum[i] = 0.0f; sSq[i] = 0.0f; }
  __syncthreads();

#pragma unroll
  for (int nt = 0; nt < NT; ++nt) {
    const int n = nt * 16 + l16;
    const float bn = bias[n];
    float ls = 0.0f, lss = 0.0f;
#pragma unroll
    for (int m = 0; m < 2; ++m) {
      const int mbase = (mt0 + m) * 16 + half16 * 8;
#pragma unroll
      for (int r = 0; r < 8; ++r) {
        float v = acc[m][nt][r] + bn;
        D[(long)(mbase + r) * Nout + n] = v;
        ls += v; lss += v * v;
      }
    }
    atomicAdd(&sSum[n], ls);
    atomicAdd(&sSq[n], lss);
  }
  __syncthreads();
  if (threadIdx.x < Nout) {
    atomicAdd(&sum[threadIdx.x],   sSum[threadIdx.x]);
    atomicAdd(&sumsq[threadIdx.x], sSq[threadIdx.x]);
  }
}

// Fold mean/var/gamma/beta into per-channel scale/shift.
__global__ void bn_finalize(const float* __restrict__ sum, const float* __restrict__ sumsq,
                            const float* __restrict__ gamma, const float* __restrict__ beta,
                            float* __restrict__ scale, float* __restrict__ shift,
                            int Nout, float invM)
{
  int nidx = blockIdx.x * blockDim.x + threadIdx.x;
  if (nidx >= Nout) return;
  float mean = sum[nidx] * invM;
  float var  = sumsq[nidx] * invM - mean * mean;
  float rstd = rsqrtf(var + 1e-5f);
  float sc = gamma[nidx] * rstd;
  scale[nidx] = sc;
  shift[nidx] = beta[nidx] - mean * sc;
}

// BN + ReLU -> f16 activations for next layer.
__global__ void bn_relu_to_h(const float* __restrict__ X,
                             const float* __restrict__ scale, const float* __restrict__ shift,
                             _Float16* __restrict__ Y, long total, int Nout)
{
  long i = (long)blockIdx.x * blockDim.x + threadIdx.x;
  if (i >= total) return;
  int n = (int)(i % Nout);
  float y = fmaxf(0.0f, X[i] * scale[n] + shift[n]);
  Y[i] = (_Float16)y;
}

// Last layer: BN + ReLU + max over K, scatter into both output copies.
__global__ void bn_relu_maxk(const float* __restrict__ X,
                             const float* __restrict__ scale, const float* __restrict__ shift,
                             float* __restrict__ out, int K, int Nout,
                             int coff, long copyDelta, int total)
{
  int i = blockIdx.x * blockDim.x + threadIdx.x;
  if (i >= total) return;
  int n  = i % Nout;
  int bs = i / Nout;
  float sc = scale[n], sh = shift[n];
  const float* row = X + (long)bs * K * Nout + n;
  float m = 0.0f;   // relu outputs are >= 0
  for (int k = 0; k < K; ++k) {
    float y = fmaxf(0.0f, row[(long)k * Nout] * sc + sh);
    m = fmaxf(m, y);
  }
  int b = bs / S_, s = bs % S_;
  long o = ((long)b * CTOT_ + (coff + n)) * S_ + s;
  out[o] = m;
  out[o + copyDelta] = m;
}

// ---------------------------------------------------------------------------
extern "C" void kernel_launch(void* const* d_in, const int* in_sizes, int n_in,
                              void* d_out, int out_size, void* d_ws, size_t ws_size,
                              hipStream_t stream)
{
  (void)in_sizes; (void)out_size; (void)ws_size;
  const float* xyz    = (const float*)d_in[0];
  const float* points = (const float*)d_in[1];
  const float* color  = (const float*)d_in[2];
  const int* fps_init1 = (const int*)d_in[n_in - 2];
  const int* fps_init2 = (const int*)d_in[n_in - 1];

  // Output offsets (floats): new_xyz | feats | new_color | feats(copy)
  const long OFF0 = 0;
  const long OFF1 = (long)B_ * 3 * S_;                 // 24576
  const long OFF2 = OFF1 + (long)B_ * CTOT_ * S_;      // 2646016
  const long OFF3 = OFF2 + (long)B_ * 3 * S_;          // 2670592
  float* out = (float*)d_out;

  // ---- workspace carve-up ----
  char* wsb = (char*)d_ws;
  size_t o = 0;
  auto take = [&](size_t bytes) -> char* {
    char* p = wsb + o;
    o = (o + bytes + 255) & ~(size_t)255;
    return p;
  };
  const long BS   = (long)B_ * S_;
  const long MAXM = BS * 64;
  int*      fpsIdx1 = (int*)take(BS * 4);
  int*      fpsIdx2 = (int*)take(BS * 4);
  float*    nxyz_ws = (float*)take(BS * 3 * 4);
  float*    ncol_ws = (float*)take(BS * 3 * 4);
  int*      gidx    = (int*)take(BS * 64 * 4);
  float*    gsq     = (float*)take(BS * 64 * 4);
  _Float16* actA    = (_Float16*)take((size_t)MAXM * 96 * 2);
  _Float16* actB    = (_Float16*)take((size_t)MAXM * 96 * 2);
  float*    preBN   = (float*)take((size_t)MAXM * 128 * 4);
  _Float16* Wh      = (_Float16*)take(128 * 96 * 2);
  float*    sum     = (float*)take(128 * 4);
  float*    sumsq   = (float*)take(128 * 4);
  float*    scale   = (float*)take(128 * 4);
  float*    shift   = (float*)take(128 * 4);

  // ---- FPS (2 runs) + gather sampled xyz/color ----
  fps_kernel<<<dim3(B_, 2), 1024, 0, stream>>>(xyz, fps_init1, fps_init2, fpsIdx1, fpsIdx2);
  gather_new<<<(unsigned)((BS + 255) / 256), 256, 0, stream>>>(
      xyz, color, fpsIdx1, fpsIdx2, nxyz_ws, ncol_ws, out + OFF0, out + OFF2);

  // ---- per-radius MSG branches (only the `gp` path reaches the outputs) ----
  const int   Ks[3]   = {16, 32, 64};
  const float r2s[3]  = {0.1f * 0.1f, 0.2f * 0.2f, 0.4f * 0.4f};
  const int   coff[3] = {0, 64, 192};
  struct LC { int cinPad, cout; };
  const LC cfg[3][3] = {
    {{96, 32}, {32, 32}, {32, 64}},
    {{96, 64}, {64, 64}, {64, 128}},
    {{96, 64}, {64, 96}, {96, 128}},
  };

  for (int r = 0; r < 3; ++r) {
    const int K = Ks[r];
    const long M = BS * K;

    ball_query<<<(unsigned)(BS / 8), 256, 0, stream>>>(xyz, nxyz_ws, gidx, gsq, r2s[r], K);
    build_gp<<<(unsigned)((M * 96 + 255) / 256), 256, 0, stream>>>(
        points, xyz, nxyz_ws, gidx, gsq, actA, K, M);

    _Float16* cur = actA;
    _Float16* nxt = actB;
    for (int j = 0; j < 3; ++j) {
      const float* W  = (const float*)d_in[4 + (r * 3 + j) * 4 + 0];
      const float* bs = (const float*)d_in[4 + (r * 3 + j) * 4 + 1];
      const float* gm = (const float*)d_in[4 + (r * 3 + j) * 4 + 2];
      const float* bt = (const float*)d_in[4 + (r * 3 + j) * 4 + 3];
      const int cinPad  = cfg[r][j].cinPad;
      const int cinReal = (j == 0) ? 71 : cinPad;
      const int cout    = cfg[r][j].cout;

      conv_weights<<<(unsigned)((cout * cinPad + 255) / 256), 256, 0, stream>>>(
          W, Wh, cout, cinReal, cinPad);
      zero_stats<<<1, 256, 0, stream>>>(sum, sumsq, 128);

      dim3 gg((unsigned)(M / 256));
      switch (cout) {
        case 32:
          gemm_bias_stats<2><<<gg, 256, 0, stream>>>(cur, Wh, bs, preBN, sum, sumsq, (int)M, cinPad);
          break;
        case 64:
          gemm_bias_stats<4><<<gg, 256, 0, stream>>>(cur, Wh, bs, preBN, sum, sumsq, (int)M, cinPad);
          break;
        case 96:
          gemm_bias_stats<6><<<gg, 256, 0, stream>>>(cur, Wh, bs, preBN, sum, sumsq, (int)M, cinPad);
          break;
        default:
          gemm_bias_stats<8><<<gg, 256, 0, stream>>>(cur, Wh, bs, preBN, sum, sumsq, (int)M, cinPad);
          break;
      }
      bn_finalize<<<1, 128, 0, stream>>>(sum, sumsq, gm, bt, scale, shift, cout,
                                         1.0f / (float)M);
      if (j < 2) {
        long tot = M * cout;
        bn_relu_to_h<<<(unsigned)((tot + 255) / 256), 256, 0, stream>>>(
            preBN, scale, shift, nxt, tot, cout);
        _Float16* t = cur; cur = nxt; nxt = t;
      } else {
        int tot = (int)(BS * cout);
        bn_relu_maxk<<<(unsigned)((tot + 255) / 256), 256, 0, stream>>>(
            preBN, scale, shift, out + OFF1, K, cout, coff[r], OFF3 - OFF1, tot);
      }
    }
  }
}
